// MPNN_88828513616435
// MI455X (gfx1250) — matrix operand order repaired and checked by reading, not compiled, not run on hardware
//
#include <hip/hip_runtime.h>
#include <hip/hip_bf16.h>

#define N_NODES 100000
#define N_EDGES 1600000
#define DN 64
#define DE 64

typedef __attribute__((ext_vector_type(16))) __bf16 v16bf;
typedef __attribute__((ext_vector_type(8)))  float  v8f;

// ---------------------------------------------------------------------------
// A-fragment (16x32 bf16, MxK) built from a 64-wide f32 source row.
// Lane l (<16):  M=l, elems 0..7 = K 0..7,  elems 8..15 = K 16..23
// Lane l (>=16): M=l-16, elems 0..7 = K 8..15, elems 8..15 = K 24..31
// `half` selects K-block 0..31 vs 32..63 of the 64-wide source.
// ---------------------------------------------------------------------------
__device__ __forceinline__ v16bf a_frag_f32(const float* __restrict__ src,
                                            int half, int lane) {
  const int c0 = (lane & 16) ? 8 : 0;
  const float* p = src + half * 32;
  v16bf a;
#pragma unroll
  for (int j = 0; j < 8; ++j) a[j]     = (__bf16)p[c0 + j];
#pragma unroll
  for (int j = 0; j < 8; ++j) a[8 + j] = (__bf16)p[16 + c0 + j];
  return a;
}

__device__ __forceinline__ v16bf a_frag_lds(const __bf16* __restrict__ row,
                                            int half, int lane) {
  const int c0 = (lane & 16) ? 8 : 0;
  const __bf16* p = row + half * 32;
  v16bf a;
#pragma unroll
  for (int j = 0; j < 8; ++j) a[j]     = p[c0 + j];
#pragma unroll
  for (int j = 0; j < 8; ++j) a[8 + j] = p[16 + c0 + j];
  return a;
}

// ---------------------------------------------------------------------------
// Pre-swizzle weight matrix W[k][n] (row-major f32, n-stride 64) into LDS as
// per-lane contiguous bf16 B-fragments. Fragment f = kb*4 + nt:
//   lane l: N = nt*16 + (l&15), elem t: K = kb*32 + ((l&16)?8:0) + (t&7) + ((t&8)?16:0)
// Storage: dst[(f*32 + lane)*16 + t]  -> each lane reads 32 contiguous bytes.
// ---------------------------------------------------------------------------
__device__ __forceinline__ void fill_bfrags(__bf16* __restrict__ dst,
                                            const float* __restrict__ W,
                                            int kblocks) {
  const int total = kblocks * 4 * 32 * 16;
  for (int idx = threadIdx.x; idx < total; idx += blockDim.x) {
    const int t    = idx & 15;
    const int lane = (idx >> 4) & 31;
    const int f    = idx >> 9;
    const int nt   = f & 3;
    const int kb   = f >> 2;
    const int n    = nt * 16 + (lane & 15);
    const int k    = kb * 32 + ((lane & 16) ? 8 : 0) + (t & 7) + ((t & 8) ? 16 : 0);
    dst[idx] = (__bf16)W[k * 64 + n];
  }
}

__device__ __forceinline__ v16bf load_bfrag(const __bf16* __restrict__ base,
                                            int f, int lane) {
  return *(const v16bf*)(base + ((f * 32 + lane) << 4));
}

// Store a 16x64 f32 C-tile (4 accumulators) into LDS as bf16 row-major [16][64]
__device__ __forceinline__ void store_c_tile(__bf16* __restrict__ hs,
                                             const v8f* acc, int lane) {
  const int mb = (lane & 16) ? 8 : 0;
  const int nl = lane & 15;
#pragma unroll
  for (int nt = 0; nt < 4; ++nt) {
    const int n = nt * 16 + nl;
#pragma unroll
    for (int r = 0; r < 8; ++r) hs[(mb + r) * 64 + n] = (__bf16)acc[nt][r];
  }
}

__device__ __forceinline__ v8f splat8(float v) {
  v8f c;
#pragma unroll
  for (int j = 0; j < 8; ++j) c[j] = v;
  return c;
}

__device__ __forceinline__ void relu8(v8f& c) {
#pragma unroll
  for (int j = 0; j < 8; ++j) c[j] = c[j] > 0.f ? c[j] : 0.f;
}

// ---------------------------------------------------------------------------
__global__ void zero_kernel(float* __restrict__ p, size_t n) {
  size_t i = (size_t)blockIdx.x * blockDim.x + threadIdx.x;
  const size_t stride = (size_t)gridDim.x * blockDim.x;
  for (; i < n; i += stride) p[i] = 0.f;
}

// ---------------------------------------------------------------------------
// Edge block: e_new = relu(relu([e|x_s|x_r]We1+b1)We2+b2)We3+b3, then
// atomic scatter-add into agg[receiver].
// One wave per 16-edge tile; weights staged once per block in LDS.
// ---------------------------------------------------------------------------
__global__ void __launch_bounds__(256) edge_kernel(
    const float* __restrict__ x, const float* __restrict__ e,
    const int* __restrict__ senders, const int* __restrict__ receivers,
    const float* __restrict__ We1, const float* __restrict__ be1,
    const float* __restrict__ We2, const float* __restrict__ be2,
    const float* __restrict__ We3, const float* __restrict__ be3,
    float* __restrict__ agg) {
  __shared__ __align__(32) __bf16 sB1[24 * 512];   // 192x64 -> 6 kb x 4 nt
  __shared__ __align__(32) __bf16 sB2[8 * 512];    // 64x64  -> 2 kb x 4 nt
  __shared__ __align__(32) __bf16 sB3[8 * 512];
  __shared__ __align__(32) __bf16 sH[8][16 * 64];  // per-wave activation tile

  fill_bfrags(sB1, We1, 6);
  fill_bfrags(sB2, We2, 2);
  fill_bfrags(sB3, We3, 2);
  __syncthreads();

  const int lane  = threadIdx.x & 31;
  const int wave  = threadIdx.x >> 5;
  const int row_l = lane & 15;
  __bf16* hs = &sH[wave][0];

  const int tiles  = N_EDGES / 16;
  const int nwaves = (gridDim.x * blockDim.x) >> 5;
  const int wid    = ((blockIdx.x * blockDim.x) + threadIdx.x) >> 5;

  for (int tile = wid; tile < tiles; tile += nwaves) {
    const int erow = tile * 16 + row_l;
    const float* esrc = e + (size_t)erow * DE;
    const float* xs   = x + (size_t)senders[erow] * DN;
    const float* xr   = x + (size_t)receivers[erow] * DN;

    // ---- layer 1: [16x192] x [192x64] ----
    v16bf a1[6];
    a1[0] = a_frag_f32(esrc, 0, lane);  a1[1] = a_frag_f32(esrc, 1, lane);
    a1[2] = a_frag_f32(xs, 0, lane);    a1[3] = a_frag_f32(xs, 1, lane);
    a1[4] = a_frag_f32(xr, 0, lane);    a1[5] = a_frag_f32(xr, 1, lane);

    v8f acc[4];
#pragma unroll
    for (int nt = 0; nt < 4; ++nt) {
      v8f c = splat8(be1[nt * 16 + row_l]);
#pragma unroll
      for (int kb = 0; kb < 6; ++kb) {
        v16bf b = load_bfrag(sB1, kb * 4 + nt, lane);
        c = __builtin_amdgcn_wmma_f32_16x16x32_bf16(false, a1[kb], false, b,
                                                    (short)0, c, false, false);
      }
      relu8(c);
      acc[nt] = c;
    }
    store_c_tile(hs, acc, lane);

    // ---- layer 2: [16x64] x [64x64] ----
    v16bf a2[2];
    a2[0] = a_frag_lds(hs + row_l * 64, 0, lane);
    a2[1] = a_frag_lds(hs + row_l * 64, 1, lane);
#pragma unroll
    for (int nt = 0; nt < 4; ++nt) {
      v8f c = splat8(be2[nt * 16 + row_l]);
#pragma unroll
      for (int kb = 0; kb < 2; ++kb) {
        v16bf b = load_bfrag(sB2, kb * 4 + nt, lane);
        c = __builtin_amdgcn_wmma_f32_16x16x32_bf16(false, a2[kb], false, b,
                                                    (short)0, c, false, false);
      }
      relu8(c);
      acc[nt] = c;
    }
    store_c_tile(hs, acc, lane);

    // ---- layer 3 (no relu) ----
    a2[0] = a_frag_lds(hs + row_l * 64, 0, lane);
    a2[1] = a_frag_lds(hs + row_l * 64, 1, lane);
#pragma unroll
    for (int nt = 0; nt < 4; ++nt) {
      v8f c = splat8(be3[nt * 16 + row_l]);
#pragma unroll
      for (int kb = 0; kb < 2; ++kb) {
        v16bf b = load_bfrag(sB3, kb * 4 + nt, lane);
        c = __builtin_amdgcn_wmma_f32_16x16x32_bf16(false, a2[kb], false, b,
                                                    (short)0, c, false, false);
      }
      acc[nt] = c;
    }

    // ---- scatter-add e_new into agg (L2-resident atomics) ----
    const int mb = (lane & 16) ? 8 : 0;
    const int nl = lane & 15;
#pragma unroll
    for (int r = 0; r < 8; ++r) {
      const int recv = receivers[tile * 16 + mb + r];
      float* dst = agg + (size_t)recv * DE;
#pragma unroll
      for (int nt = 0; nt < 4; ++nt)
        atomicAdd(dst + nt * 16 + nl, acc[nt][r]);
    }
  }
}

// ---------------------------------------------------------------------------
// Node block: x_new = relu(relu([x|agg]Wn1+b1)Wn2+b2)Wn3+b3
// ---------------------------------------------------------------------------
__global__ void __launch_bounds__(256) node_kernel(
    const float* __restrict__ x, const float* __restrict__ agg,
    const float* __restrict__ Wn1, const float* __restrict__ bn1,
    const float* __restrict__ Wn2, const float* __restrict__ bn2,
    const float* __restrict__ Wn3, const float* __restrict__ bn3,
    float* __restrict__ out) {
  __shared__ __align__(32) __bf16 sB1[16 * 512];   // 128x64 -> 4 kb x 4 nt
  __shared__ __align__(32) __bf16 sB2[8 * 512];
  __shared__ __align__(32) __bf16 sB3[8 * 512];
  __shared__ __align__(32) __bf16 sH[8][16 * 64];

  fill_bfrags(sB1, Wn1, 4);
  fill_bfrags(sB2, Wn2, 2);
  fill_bfrags(sB3, Wn3, 2);
  __syncthreads();

  const int lane  = threadIdx.x & 31;
  const int wave  = threadIdx.x >> 5;
  const int row_l = lane & 15;
  __bf16* hs = &sH[wave][0];

  const int tiles  = N_NODES / 16;
  const int nwaves = (gridDim.x * blockDim.x) >> 5;
  const int wid    = ((blockIdx.x * blockDim.x) + threadIdx.x) >> 5;

  for (int tile = wid; tile < tiles; tile += nwaves) {
    const int nrow = tile * 16 + row_l;
    const float* xsrc = x   + (size_t)nrow * DN;
    const float* asrc = agg + (size_t)nrow * DE;

    v16bf a1[4];
    a1[0] = a_frag_f32(xsrc, 0, lane);  a1[1] = a_frag_f32(xsrc, 1, lane);
    a1[2] = a_frag_f32(asrc, 0, lane);  a1[3] = a_frag_f32(asrc, 1, lane);

    v8f acc[4];
#pragma unroll
    for (int nt = 0; nt < 4; ++nt) {
      v8f c = splat8(bn1[nt * 16 + row_l]);
#pragma unroll
      for (int kb = 0; kb < 4; ++kb) {
        v16bf b = load_bfrag(sB1, kb * 4 + nt, lane);
        c = __builtin_amdgcn_wmma_f32_16x16x32_bf16(false, a1[kb], false, b,
                                                    (short)0, c, false, false);
      }
      relu8(c);
      acc[nt] = c;
    }
    store_c_tile(hs, acc, lane);

    v16bf a2[2];
    a2[0] = a_frag_lds(hs + row_l * 64, 0, lane);
    a2[1] = a_frag_lds(hs + row_l * 64, 1, lane);
#pragma unroll
    for (int nt = 0; nt < 4; ++nt) {
      v8f c = splat8(bn2[nt * 16 + row_l]);
#pragma unroll
      for (int kb = 0; kb < 2; ++kb) {
        v16bf b = load_bfrag(sB2, kb * 4 + nt, lane);
        c = __builtin_amdgcn_wmma_f32_16x16x32_bf16(false, a2[kb], false, b,
                                                    (short)0, c, false, false);
      }
      relu8(c);
      acc[nt] = c;
    }
    store_c_tile(hs, acc, lane);

    a2[0] = a_frag_lds(hs + row_l * 64, 0, lane);
    a2[1] = a_frag_lds(hs + row_l * 64, 1, lane);
#pragma unroll
    for (int nt = 0; nt < 4; ++nt) {
      v8f c = splat8(bn3[nt * 16 + row_l]);
#pragma unroll
      for (int kb = 0; kb < 2; ++kb) {
        v16bf b = load_bfrag(sB3, kb * 4 + nt, lane);
        c = __builtin_amdgcn_wmma_f32_16x16x32_bf16(false, a2[kb], false, b,
                                                    (short)0, c, false, false);
      }
      acc[nt] = c;
    }

    // write x_new
    const int mb = (lane & 16) ? 8 : 0;
    const int nl = lane & 15;
#pragma unroll
    for (int r = 0; r < 8; ++r) {
      float* dst = out + (size_t)(tile * 16 + mb + r) * DN;
#pragma unroll
      for (int nt = 0; nt < 4; ++nt) dst[nt * 16 + nl] = acc[nt][r];
    }
  }
}

// ---------------------------------------------------------------------------
extern "C" void kernel_launch(void* const* d_in, const int* in_sizes, int n_in,
                              void* d_out, int out_size, void* d_ws, size_t ws_size,
                              hipStream_t stream) {
  const float* x         = (const float*)d_in[0];
  const float* e         = (const float*)d_in[1];
  const int*   senders   = (const int*)d_in[2];
  const int*   receivers = (const int*)d_in[3];
  const float* We1 = (const float*)d_in[4];  const float* be1 = (const float*)d_in[5];
  const float* We2 = (const float*)d_in[6];  const float* be2 = (const float*)d_in[7];
  const float* We3 = (const float*)d_in[8];  const float* be3 = (const float*)d_in[9];
  const float* Wn1 = (const float*)d_in[10]; const float* bn1 = (const float*)d_in[11];
  const float* Wn2 = (const float*)d_in[12]; const float* bn2 = (const float*)d_in[13];
  const float* Wn3 = (const float*)d_in[14]; const float* bn3 = (const float*)d_in[15];

  float* agg = (float*)d_ws;                      // [N_NODES, DE] f32 scratch
  (void)in_sizes; (void)n_in; (void)out_size; (void)ws_size;

  zero_kernel<<<2048, 256, 0, stream>>>(agg, (size_t)N_NODES * DE);
  edge_kernel<<<1024, 256, 0, stream>>>(x, e, senders, receivers,
                                        We1, be1, We2, be2, We3, be3, agg);
  node_kernel<<<512, 256, 0, stream>>>(x, agg, Wn1, bn1, Wn2, bn2, Wn3, bn3,
                                       (float*)d_out);
}
